// SelfAttentionLayer_4011499455207
// MI455X (gfx1250) — matrix operand order repaired and checked by reading, not compiled, not run on hardware
//
#include <hip/hip_runtime.h>
#include <hip/hip_bf16.h>

// Problem constants (from the reference)
#define BB 8
#define SS 2048
#define FF 512
#define DD 512

typedef __attribute__((ext_vector_type(16))) __bf16 v16bf;
typedef __attribute__((ext_vector_type(8)))  float  v8f;

#define LDS_STRIDE 36  // even -> aligned 32-bit pair loads; padded vs 32 to spread banks

__device__ __forceinline__ unsigned short f32_to_bf16(float f) {
    union { float f; unsigned int u; } x;
    x.f = f;
    unsigned int u = x.u;
    u += 0x7FFFu + ((u >> 16) & 1u);   // round-to-nearest-even
    return (unsigned short)(u >> 16);
}

union BfFrag { v16bf v; unsigned int u[8]; };

// A-matrix fragment (16x32 bf16): rowbase = subtile origin row in LDS tile
__device__ __forceinline__ BfFrag load_frag_A(const unsigned short* __restrict__ As,
                                              int rowbase, int lane) {
    const int half = lane >> 4;
    const int l15  = lane & 15;
    const unsigned short* arow = As + (rowbase + l15) * LDS_STRIDE;
    BfFrag a;
#pragma unroll
    for (int v = 0; v < 8; ++v) {
        const int ka = 2 * (v & 3) + 8 * half + ((v >= 4) ? 16 : 0);
        a.u[v] = *(const unsigned int*)(arow + ka);
    }
    return a;
}

// B-matrix fragment (32x16 bf16), B staged N-major in LDS (Bt[n][k])
__device__ __forceinline__ BfFrag load_frag_B(const unsigned short* __restrict__ Bt,
                                              int rowbase, int lane) {
    const int half = lane >> 4;
    const int l15  = lane & 15;
    const unsigned short* brow = Bt + (rowbase + l15) * LDS_STRIDE;
    BfFrag b;
#pragma unroll
    for (int v = 0; v < 8; ++v) {
        const int kb = 2 * (v & 3) + 8 * half + ((v >= 4) ? 16 : 0);
        b.u[v] = *(const unsigned int*)(brow + kb);
    }
    return b;
}

#define WMMA_BF16(a, b, c) \
    __builtin_amdgcn_wmma_f32_16x16x32_bf16(false, (a).v, false, (b).v, (short)0, (c), false, false)

// ---------------------------------------------------------------------------
// Phase 1: q/k/v projections.  Y(bf16)[B*S, D] = X(f32)[B*S, F] @ W(f32)[F, D]
// Block tile 64x64, 4 waves, each wave 32x32 (2x2 WMMA frags).
// grid = (D/64, B*S/64, 3), block = 128
// ---------------------------------------------------------------------------
__global__ __launch_bounds__(128) void k_proj(
    const float* __restrict__ qx, const float* __restrict__ kx, const float* __restrict__ vx,
    const float* __restrict__ Wq, const float* __restrict__ Wk, const float* __restrict__ Wv,
    unsigned short* __restrict__ qb, unsigned short* __restrict__ kb, unsigned short* __restrict__ vb)
{
    __shared__ unsigned short As[64 * LDS_STRIDE];  // 64 rows (M) x 32 cols (K)
    __shared__ unsigned short Bt[64 * LDS_STRIDE];  // 64 rows (N) x 32 cols (K)

    const int z = blockIdx.z;
    const float* X = (z == 0) ? qx : ((z == 1) ? kx : vx);
    const float* W = (z == 0) ? Wq : ((z == 1) ? Wk : Wv);
    unsigned short* Y = (z == 0) ? qb : ((z == 1) ? kb : vb);

    const int n0 = blockIdx.x * 64;   // output cols (D)
    const int r0 = blockIdx.y * 64;   // output rows (B*S flattened)
    const int tid  = threadIdx.x;
    const int lane = tid & 31;
    const int wave = tid >> 5;
    const int wm = wave >> 1, wn = wave & 1;   // wave quadrant of 64x64

    v8f acc00 = {}, acc01 = {}, acc10 = {}, acc11 = {};

    for (int k0 = 0; k0 < FF; k0 += 32) {
        // Stage A: 64x32, direct (convert f32->bf16)
#pragma unroll
        for (int i = 0; i < 16; ++i) {
            const int idx = tid + i * 128;
            const int r = idx >> 5, c = idx & 31;
            As[r * LDS_STRIDE + c] = f32_to_bf16(X[(size_t)(r0 + r) * FF + (k0 + c)]);
        }
        // Stage B: W[K=32][N=64] row-major -> Bt[n][k]; consecutive threads read consecutive n
#pragma unroll
        for (int i = 0; i < 16; ++i) {
            const int idx = tid + i * 128;
            const int k = idx >> 6, n = idx & 63;
            Bt[n * LDS_STRIDE + k] = f32_to_bf16(W[(size_t)(k0 + k) * DD + (n0 + n)]);
        }
        __syncthreads();

        const BfFrag a0 = load_frag_A(As, wm * 32 +  0, lane);
        const BfFrag a1 = load_frag_A(As, wm * 32 + 16, lane);
        const BfFrag b0 = load_frag_B(Bt, wn * 32 +  0, lane);
        const BfFrag b1 = load_frag_B(Bt, wn * 32 + 16, lane);
        acc00 = WMMA_BF16(a0, b0, acc00);
        acc01 = WMMA_BF16(a0, b1, acc01);
        acc10 = WMMA_BF16(a1, b0, acc10);
        acc11 = WMMA_BF16(a1, b1, acc11);
        __syncthreads();
    }

    const int half = lane >> 4;
    const int l15  = lane & 15;
    const v8f* accs[4] = { &acc00, &acc01, &acc10, &acc11 };
#pragma unroll
    for (int mi = 0; mi < 2; ++mi) {
#pragma unroll
        for (int ni = 0; ni < 2; ++ni) {
            const v8f& a = *accs[mi * 2 + ni];
            const int col = n0 + wn * 32 + ni * 16 + l15;
#pragma unroll
            for (int v = 0; v < 8; ++v) {
                const int row = r0 + wm * 32 + mi * 16 + v + 8 * half;
                Y[(size_t)row * DD + col] = f32_to_bf16(a[v]);
            }
        }
    }
}

// ---------------------------------------------------------------------------
// Phase 2: logits[b,i,j] = scale * (q[b,i,:] . k[b,j,:]) - 1e9 * mask[b,i,j]
// grid = (S/64, S/64, B), block = 128.  Writes f32 logits into p_out.
// ---------------------------------------------------------------------------
__global__ __launch_bounds__(128) void k_scores(
    const unsigned short* __restrict__ qb, const unsigned short* __restrict__ kb,
    const float* __restrict__ mask, float* __restrict__ p_out)
{
    __shared__ unsigned short As[64 * LDS_STRIDE];
    __shared__ unsigned short Bt[64 * LDS_STRIDE];

    const int b  = blockIdx.z;
    const int j0 = blockIdx.x * 64;   // key index (cols)
    const int i0 = blockIdx.y * 64;   // query index (rows)
    const unsigned short* q = qb + (size_t)b * SS * DD;
    const unsigned short* k = kb + (size_t)b * SS * DD;

    const int tid  = threadIdx.x;
    const int lane = tid & 31;
    const int wave = tid >> 5;
    const int wm = wave >> 1, wn = wave & 1;

    v8f acc00 = {}, acc01 = {}, acc10 = {}, acc11 = {};

    for (int k0 = 0; k0 < DD; k0 += 32) {
#pragma unroll
        for (int i = 0; i < 16; ++i) {
            const int idx = tid + i * 128;
            const int r = idx >> 5, c = idx & 31;
            As[r * LDS_STRIDE + c] = q[(size_t)(i0 + r) * DD + (k0 + c)];
            // B = k^T : Bt[n][kd] = k[j0+n][k0+kd]  -> direct row-major copy of k rows
            Bt[r * LDS_STRIDE + c] = k[(size_t)(j0 + r) * DD + (k0 + c)];
        }
        __syncthreads();

        const BfFrag a0 = load_frag_A(As, wm * 32 +  0, lane);
        const BfFrag a1 = load_frag_A(As, wm * 32 + 16, lane);
        const BfFrag b0 = load_frag_B(Bt, wn * 32 +  0, lane);
        const BfFrag b1 = load_frag_B(Bt, wn * 32 + 16, lane);
        acc00 = WMMA_BF16(a0, b0, acc00);
        acc01 = WMMA_BF16(a0, b1, acc01);
        acc10 = WMMA_BF16(a1, b0, acc10);
        acc11 = WMMA_BF16(a1, b1, acc11);
        __syncthreads();
    }

    const float scale = 0.044194173824159216f;  // 1/sqrt(512)
    const int half = lane >> 4;
    const int l15  = lane & 15;
    const v8f* accs[4] = { &acc00, &acc01, &acc10, &acc11 };
#pragma unroll
    for (int mi = 0; mi < 2; ++mi) {
#pragma unroll
        for (int ni = 0; ni < 2; ++ni) {
            const v8f& a = *accs[mi * 2 + ni];
            const int j = j0 + wn * 32 + ni * 16 + l15;
#pragma unroll
            for (int v = 0; v < 8; ++v) {
                const int i = i0 + wm * 32 + mi * 16 + v + 8 * half;
                const size_t idx = (size_t)b * SS * SS + (size_t)i * SS + j;
                p_out[idx] = a[v] * scale + mask[idx] * (-1e9f);
            }
        }
    }
}

// ---------------------------------------------------------------------------
// Phase 3: row softmax over S=2048; in-place f32 in p_out, bf16 copy to pb.
// grid = (S, B), block = 256 (8 elements/thread)
// ---------------------------------------------------------------------------
__global__ __launch_bounds__(256) void k_softmax(float* __restrict__ p,
                                                 unsigned short* __restrict__ pb)
{
    const int b = blockIdx.y;
    const int i = blockIdx.x;
    float* row           = p  + (size_t)b * SS * SS + (size_t)i * SS;
    unsigned short* rowb = pb + (size_t)b * SS * SS + (size_t)i * SS;

    const int tid = threadIdx.x;
    __shared__ float red[256];

    float vals[8];
    float m = -3.4e38f;
#pragma unroll
    for (int e = 0; e < 8; ++e) {
        vals[e] = row[tid + e * 256];
        m = fmaxf(m, vals[e]);
    }
    red[tid] = m;
    __syncthreads();
    for (int s = 128; s > 0; s >>= 1) {
        if (tid < s) red[tid] = fmaxf(red[tid], red[tid + s]);
        __syncthreads();
    }
    m = red[0];
    __syncthreads();

    float sum = 0.0f;
#pragma unroll
    for (int e = 0; e < 8; ++e) {
        vals[e] = __expf(vals[e] - m);
        sum += vals[e];
    }
    red[tid] = sum;
    __syncthreads();
    for (int s = 128; s > 0; s >>= 1) {
        if (tid < s) red[tid] += red[tid + s];
        __syncthreads();
    }
    const float inv = 1.0f / red[0];

#pragma unroll
    for (int e = 0; e < 8; ++e) {
        const float pe = vals[e] * inv;
        row[tid + e * 256]  = pe;
        rowb[tid + e * 256] = f32_to_bf16(pe);
    }
}

// ---------------------------------------------------------------------------
// Phase 4: h[b,i,n] = sum_j p[b,i,j] * v[b,j,n]
// grid = (D/64, S/64, B), block = 128
// ---------------------------------------------------------------------------
__global__ __launch_bounds__(128) void k_hout(
    const unsigned short* __restrict__ pb, const unsigned short* __restrict__ vb,
    float* __restrict__ h)
{
    __shared__ unsigned short As[64 * LDS_STRIDE];
    __shared__ unsigned short Bt[64 * LDS_STRIDE];

    const int b  = blockIdx.z;
    const int n0 = blockIdx.x * 64;   // D cols
    const int i0 = blockIdx.y * 64;   // S rows
    const unsigned short* P = pb + (size_t)b * SS * SS;
    const unsigned short* V = vb + (size_t)b * SS * DD;

    const int tid  = threadIdx.x;
    const int lane = tid & 31;
    const int wave = tid >> 5;
    const int wm = wave >> 1, wn = wave & 1;

    v8f acc00 = {}, acc01 = {}, acc10 = {}, acc11 = {};

    for (int k0 = 0; k0 < SS; k0 += 32) {
#pragma unroll
        for (int i = 0; i < 16; ++i) {
            const int idx = tid + i * 128;
            const int r = idx >> 5, c = idx & 31;
            As[r * LDS_STRIDE + c] = P[(size_t)(i0 + r) * SS + (k0 + c)];
        }
        // V is [S(k), D(n)] row-major (K x N): transpose into Bt[n][k]
#pragma unroll
        for (int i = 0; i < 16; ++i) {
            const int idx = tid + i * 128;
            const int k = idx >> 6, n = idx & 63;
            Bt[n * LDS_STRIDE + k] = V[(size_t)(k0 + k) * DD + (n0 + n)];
        }
        __syncthreads();

        const BfFrag a0 = load_frag_A(As, wm * 32 +  0, lane);
        const BfFrag a1 = load_frag_A(As, wm * 32 + 16, lane);
        const BfFrag b0 = load_frag_B(Bt, wn * 32 +  0, lane);
        const BfFrag b1 = load_frag_B(Bt, wn * 32 + 16, lane);
        acc00 = WMMA_BF16(a0, b0, acc00);
        acc01 = WMMA_BF16(a0, b1, acc01);
        acc10 = WMMA_BF16(a1, b0, acc10);
        acc11 = WMMA_BF16(a1, b1, acc11);
        __syncthreads();
    }

    const int half = lane >> 4;
    const int l15  = lane & 15;
    const v8f* accs[4] = { &acc00, &acc01, &acc10, &acc11 };
#pragma unroll
    for (int mi = 0; mi < 2; ++mi) {
#pragma unroll
        for (int ni = 0; ni < 2; ++ni) {
            const v8f& a = *accs[mi * 2 + ni];
            const int col = n0 + wn * 32 + ni * 16 + l15;
#pragma unroll
            for (int v = 0; v < 8; ++v) {
                const int row = i0 + wm * 32 + mi * 16 + v + 8 * half;
                h[(size_t)b * SS * DD + (size_t)row * DD + col] = a[v];
            }
        }
    }
}

// ---------------------------------------------------------------------------
extern "C" void kernel_launch(void* const* d_in, const int* in_sizes, int n_in,
                              void* d_out, int out_size, void* d_ws, size_t ws_size,
                              hipStream_t stream)
{
    (void)in_sizes; (void)n_in; (void)out_size; (void)ws_size;

    const float* qx   = (const float*)d_in[0];
    const float* kx   = (const float*)d_in[1];
    const float* vx   = (const float*)d_in[2];
    const float* mask = (const float*)d_in[3];
    const float* Wq   = (const float*)d_in[4];
    const float* Wk   = (const float*)d_in[5];
    const float* Wv   = (const float*)d_in[6];

    float* h_out = (float*)d_out;                         // [B,S,D]
    float* p_out = h_out + (size_t)BB * SS * DD;          // [B,S,S]

    // Workspace: bf16 q,k,v [B,S,D] + bf16 p [B,S,S]  (~112 MB)
    unsigned short* qb = (unsigned short*)d_ws;
    unsigned short* kb = qb + (size_t)BB * SS * DD;
    unsigned short* vb = kb + (size_t)BB * SS * DD;
    unsigned short* pb = vb + (size_t)BB * SS * DD;

    dim3 blk(128);

    // Phase 1: projections (3 GEMMs fused into one launch via grid.z)
    k_proj<<<dim3(DD / 64, (BB * SS) / 64, 3), blk, 0, stream>>>(
        qx, kx, vx, Wq, Wk, Wv, qb, kb, vb);

    // Phase 2: masked, scaled logits
    k_scores<<<dim3(SS / 64, SS / 64, BB), blk, 0, stream>>>(qb, kb, mask, p_out);

    // Phase 3: softmax rows (f32 in place + bf16 copy)
    k_softmax<<<dim3(SS, BB), dim3(256), 0, stream>>>(p_out, pb);

    // Phase 4: h = p @ v
    k_hout<<<dim3(DD / 64, SS / 64, BB), blk, 0, stream>>>(pb, vb, h_out);
}
